// GATLayer_7438883357554
// MI455X (gfx1250) — matrix operand order repaired and checked by reading, not compiled, not run on hardware
//
#include <hip/hip_runtime.h>
#include <hip/hip_bf16.h>

// ---------------------------------------------------------------------------
// GAT layer, N=4096, IN=256, OUT=256, HEADS=4, HEAD_DIM=64
// d_in[0]=h (f32 4096x256), d_in[1]=adj (i32 4096x4096),
// d_in[2]=W (f32 256x256), d_in[3]=a (f32 4x128); d_out = f32 4096x256
// Strategy: Wh via f16 WMMA; masked softmax aggregation as flash-attention
// with additive scores (base-2 domain), j split into segments for occupancy.
// ---------------------------------------------------------------------------

typedef __attribute__((ext_vector_type(16))) _Float16 v16h;
typedef __attribute__((ext_vector_type(8)))  _Float16 v8h;
typedef __attribute__((ext_vector_type(8)))  float    v8f;

#define GAT_N   4096
#define GAT_IN  256
#define GAT_OUT 256
#define GAT_H   4
#define GAT_D   64
#define LOG2E   1.44269504088896340736f

__device__ __forceinline__ v8f wmma_f16(v16h a, v16h b, v8f c) {
    // (neg_a, A, neg_b, B, c_mod, C, reuse_a, reuse_b)
    return __builtin_amdgcn_wmma_f32_16x16x32_f16(false, a, false, b,
                                                  (short)0, c, false, false);
}

// --- prep kernels -----------------------------------------------------------

__global__ void cvt_h_kernel(const float* __restrict__ h,
                             _Float16* __restrict__ hF, int n) {
    int i = blockIdx.x * blockDim.x + threadIdx.x;
    if (i < n) hF[i] = (_Float16)h[i];
}

__global__ void transpose_W_kernel(const float* __restrict__ W,
                                   _Float16* __restrict__ WT) {
    int i = blockIdx.x * blockDim.x + threadIdx.x;   // 0 .. 65535
    int k = i >> 8;        // row of W   (IN dim)
    int n = i & 255;       // col of W   (OUT dim)
    WT[n * GAT_IN + k] = (_Float16)W[k * GAT_OUT + n];
}

__global__ void zero_kernel(float* __restrict__ p, int n) {
    int i = blockIdx.x * blockDim.x + threadIdx.x;
    if (i < n) p[i] = 0.0f;
}

// --- phase 1: Wh = h @ W  (WMMA f16 -> f32), also WhT(f16) + s_i/s_j --------
// One wave computes a 16x64 tile of Wh. N-group (64 wide) == one head.
// Scores are pre-scaled by log2(e) so phase 2 can use bare v_exp_f32 (2^x).

__global__ __launch_bounds__(256)
void gemm1_kernel(const _Float16* __restrict__ hF,
                  const _Float16* __restrict__ WT,
                  const float* __restrict__ avec,     // (4,128)
                  _Float16* __restrict__ WhT,         // [256][4096] f16
                  float* __restrict__ siT,            // [4][4096]
                  float* __restrict__ sjT) {          // [4][4096]
    const int lane = threadIdx.x & 31;
    const int wv   = threadIdx.x >> 5;
    const int gw   = blockIdx.x * 8 + wv;            // 0..1023
    const int mt   = gw >> 2;                        // 0..255 (row tile)
    const int hd   = gw & 3;                         // head == N group
    const int m    = lane & 15;
    const int part = lane >> 4;

    v8f acc[4];
    for (int nt = 0; nt < 4; ++nt)
        for (int r = 0; r < 8; ++r) acc[nt][r] = 0.0f;

    const int row = mt * 16 + m;
    for (int k = 0; k < GAT_IN; k += 32) {
        // A fragment (16x32 f16): lane row = m, K halves {kb..kb+7, kb+16..kb+23}
        const _Float16* pa = hF + row * GAT_IN + k + part * 8;
        v8h alo = *(const v8h*)pa;
        v8h ahi = *(const v8h*)(pa + 16);
        v16h af;
        for (int t = 0; t < 8; ++t) { af[t] = alo[t]; af[8 + t] = ahi[t]; }

        for (int nt = 0; nt < 4; ++nt) {
            // B fragment (32x16): lane col = m, K halves {kb..kb+7, kb+16..kb+23} in WT
            const int ncol = hd * 64 + nt * 16 + m;
            const _Float16* pb = WT + ncol * GAT_IN + k + part * 8;
            v8h b0 = *(const v8h*)pb;
            v8h b1 = *(const v8h*)(pb + 16);
            v16h bf;
            for (int t = 0; t < 8; ++t) { bf[t] = b0[t]; bf[8 + t] = b1[t]; }
            acc[nt] = wmma_f16(af, bf, acc[nt]);
        }
    }

    // Epilogue: write WhT (f16, transposed) + per-row score partials (x log2e)
    float psi[8], psj[8];
    for (int r = 0; r < 8; ++r) { psi[r] = 0.0f; psj[r] = 0.0f; }
    for (int nt = 0; nt < 4; ++nt) {
        const int dcol = nt * 16 + m;                // 0..63 within head
        const float ai = avec[hd * 128 + dcol] * LOG2E;
        const float aj = avec[hd * 128 + 64 + dcol] * LOG2E;
        for (int r = 0; r < 8; ++r) {
            const float val = acc[nt][r];
            const int node = mt * 16 + part * 8 + r; // C layout: M = part*8+r
            WhT[(size_t)(hd * 64 + dcol) * GAT_N + node] = (_Float16)val;
            psi[r] += val * ai;
            psj[r] += val * aj;
        }
    }
    for (int r = 0; r < 8; ++r) {
        float ci = psi[r], cj = psj[r];
        for (int off = 1; off < 16; off <<= 1) {
            ci += __shfl_xor(ci, off);
            cj += __shfl_xor(cj, off);
        }
        if (m == 0) {
            const int node = mt * 16 + part * 8 + r;
            atomicAdd(&siT[hd * GAT_N + node], ci);
            atomicAdd(&sjT[hd * GAT_N + node], cj);
        }
    }
}

// --- phase 2: flash-style masked softmax + alpha @ Wh -----------------------
// Block = 16 rows; 4 waves, one head per wave; grid.y = j-segment.
// Scores already in base-2 domain; 2^x via v_exp_f32.

__global__ __launch_bounds__(128)
void gat_kernel(const int* __restrict__ adj,
                const float* __restrict__ siT,
                const float* __restrict__ sjT,
                const _Float16* __restrict__ WhT,
                float* __restrict__ out,             // used when nseg==1
                float* __restrict__ Pacc,            // [tile][seg][h][16][64]
                float* __restrict__ Pm,              // [tile][seg][h][16]
                float* __restrict__ Pl,              // [tile][seg][h][16]
                int seg_len, int nseg) {
    const int i0   = blockIdx.x * 16;
    const int seg  = blockIdx.y;
    const int hd   = threadIdx.y;                    // head (one wave each)
    const int lane = threadIdx.x;                    // wave32
    const int m    = lane & 15;
    const int part = lane >> 4;

    const float si_m = siT[hd * GAT_N + i0 + m];
    const float* sjrow = sjT + hd * GAT_N;
    const int* arow = adj + (size_t)(i0 + m) * GAT_N;

    float m_run = -1.0e30f;
    float l_run = 0.0f;
    v8f acc[4];
    for (int nt = 0; nt < 4; ++nt)
        for (int r = 0; r < 8; ++r) acc[nt][r] = 0.0f;

    const int jlo = seg * seg_len;
    const int jhi = jlo + seg_len;
    for (int j0 = jlo; j0 < jhi; j0 += 32) {
        const int jb0 = j0 + part * 8;               // P halves 0..7
        const int jb1 = j0 + 16 + part * 8;          // P halves 8..15

        // speculative prefetch of next chunk's adjacency bytes
        // (global_prefetch_b8; dropped silently if past the mapping)
        __builtin_prefetch(arow + jb0 + 32, 0, 0);

        // adjacency tile 16x32 (b128 loads)
        int4 A0 = *(const int4*)(arow + jb0);
        int4 A1 = *(const int4*)(arow + jb0 + 4);
        int4 A2 = *(const int4*)(arow + jb1);
        int4 A3 = *(const int4*)(arow + jb1 + 4);
        float4 S0 = *(const float4*)(sjrow + jb0);
        float4 S1 = *(const float4*)(sjrow + jb0 + 4);
        float4 S2 = *(const float4*)(sjrow + jb1);
        float4 S3 = *(const float4*)(sjrow + jb1 + 4);

        int   av[16] = {A0.x,A0.y,A0.z,A0.w, A1.x,A1.y,A1.z,A1.w,
                        A2.x,A2.y,A2.z,A2.w, A3.x,A3.y,A3.z,A3.w};
        float sv[16] = {S0.x,S0.y,S0.z,S0.w, S1.x,S1.y,S1.z,S1.w,
                        S2.x,S2.y,S2.z,S2.w, S3.x,S3.y,S3.z,S3.w};

        float ev[16];
        float cm = -1.0e30f;
        for (int t = 0; t < 16; ++t) {
            float x = si_m + sv[t];
            x = (x > 0.0f) ? x : 0.2f * x;           // leaky_relu(0.2)
            x = (av[t] == 0) ? -1.0e30f : x;         // mask (base-2 sentinel)
            ev[t] = x;
            cm = fmaxf(cm, x);
        }
        cm = fmaxf(cm, __shfl_xor(cm, 16));          // combine both j-halves

        // rescale only when some row's max actually moved (rare after warmup)
        if (__any(cm > m_run)) {
            const float mnew  = fmaxf(m_run, cm);
            const float scale = __builtin_amdgcn_exp2f(m_run - mnew);
            m_run = mnew;
            l_run *= scale;
            float srow[8];
            for (int r = 0; r < 8; ++r) srow[r] = __shfl(scale, part * 8 + r);
            for (int nt = 0; nt < 4; ++nt)
                for (int r = 0; r < 8; ++r) acc[nt][r] *= srow[r];
        }

        // P fragment in 16-bit A-matrix layout + row sum (p = 2^(e - m))
        v16h af;
        float psum = 0.0f;
        for (int t = 0; t < 16; ++t) {
            float p = __builtin_amdgcn_exp2f(ev[t] - m_run);
            psum += p;
            af[t] = (_Float16)p;
        }
        psum += __shfl_xor(psum, 16);
        l_run += psum;

        // V fragments from WhT and accumulate
        for (int nt = 0; nt < 4; ++nt) {
            const int d = hd * 64 + nt * 16 + m;
            const _Float16* pb = WhT + (size_t)d * GAT_N + j0 + part * 8;
            v8h b0 = *(const v8h*)pb;
            v8h b1 = *(const v8h*)(pb + 16);
            v16h bf;
            for (int t = 0; t < 8; ++t) { bf[t] = b0[t]; bf[8 + t] = b1[t]; }
            acc[nt] = wmma_f16(af, bf, acc[nt]);
        }
    }

    if (nseg == 1) {
        // normalize, ELU, store directly
        float linv[8];
        for (int r = 0; r < 8; ++r) {
            const float lr = __shfl(l_run, part * 8 + r);
            linv[r] = 1.0f / lr;
        }
        for (int nt = 0; nt < 4; ++nt) {
            for (int r = 0; r < 8; ++r) {
                float v = acc[nt][r] * linv[r];
                v = (v > 0.0f) ? v : (__expf(v) - 1.0f);  // elu
                const int node = i0 + part * 8 + r;
                out[(size_t)node * GAT_OUT + hd * 64 + nt * 16 + m] = v;
            }
        }
    } else {
        // write unnormalized partials for the combine pass
        const size_t b = (size_t)(blockIdx.x * nseg + seg) * GAT_H + hd;
        float* pa = Pacc + (b << 10);                // 16*64 floats
        for (int nt = 0; nt < 4; ++nt)
            for (int r = 0; r < 8; ++r)
                pa[(part * 8 + r) * 64 + nt * 16 + m] = acc[nt][r];
        if (lane < 16) {
            Pm[(b << 4) + lane] = m_run;             // row = lane
            Pl[(b << 4) + lane] = l_run;
        }
    }
}

// --- phase 3: merge j-segments, normalize, ELU ------------------------------

__global__ __launch_bounds__(256)
void combine_kernel(const float* __restrict__ Pacc,
                    const float* __restrict__ Pm,
                    const float* __restrict__ Pl,
                    float* __restrict__ out, int nseg) {
    const int col  = threadIdx.x;                    // 0..63
    const int hd   = threadIdx.y;                    // head
    const int tile = blockIdx.x;
    for (int row = 0; row < 16; ++row) {
        float M = -1.0e30f;
        for (int s = 0; s < nseg; ++s) {
            const size_t b = (size_t)(tile * nseg + s) * GAT_H + hd;
            M = fmaxf(M, Pm[(b << 4) + row]);
        }
        float L = 0.0f, O = 0.0f;
        for (int s = 0; s < nseg; ++s) {
            const size_t b = (size_t)(tile * nseg + s) * GAT_H + hd;
            const float w = __builtin_amdgcn_exp2f(Pm[(b << 4) + row] - M);
            L += Pl[(b << 4) + row] * w;
            O += Pacc[(b << 10) + row * 64 + col] * w;
        }
        float v = O / L;
        v = (v > 0.0f) ? v : (__expf(v) - 1.0f);     // elu
        out[(size_t)(tile * 16 + row) * GAT_OUT + hd * 64 + col] = v;
    }
}

// ---------------------------------------------------------------------------

extern "C" void kernel_launch(void* const* d_in, const int* in_sizes, int n_in,
                              void* d_out, int out_size, void* d_ws, size_t ws_size,
                              hipStream_t stream) {
    const float* h   = (const float*)d_in[0];
    const int*   adj = (const int*)d_in[1];
    const float* W   = (const float*)d_in[2];
    const float* a   = (const float*)d_in[3];
    float* out = (float*)d_out;

    char* ws = (char*)d_ws;
    _Float16* hF  = (_Float16*)(ws + 0);              // 2 MB
    _Float16* WT  = (_Float16*)(ws + 2097152);        // 128 KB
    _Float16* WhT = (_Float16*)(ws + 2228224);        // 2 MB
    float*    siT = (float*)(ws + 4325376);           // 64 KB
    float*    sjT = (float*)(ws + 4390912);           // 64 KB
    const size_t prep_bytes = 4456448;

    // pick the largest segment count whose partial buffers fit in workspace
    const int ntiles = GAT_N / 16;                    // 256
    int S = 4;
    while (S > 1) {
        size_t need = prep_bytes
                    + (size_t)S * ntiles * GAT_H * 1024 * 4   // Pacc
                    + 2 * ((size_t)S * ntiles * GAT_H * 16 * 4);
        if (need <= ws_size) break;
        S >>= 1;
    }
    float* Pacc = nullptr; float* Pm = nullptr; float* Pl = nullptr;
    if (S > 1) {
        size_t o = prep_bytes;
        Pacc = (float*)(ws + o); o += (size_t)S * ntiles * GAT_H * 1024 * 4;
        Pm   = (float*)(ws + o); o += (size_t)S * ntiles * GAT_H * 16 * 4;
        Pl   = (float*)(ws + o);
    }

    cvt_h_kernel<<<dim3((GAT_N * GAT_IN + 255) / 256), dim3(256), 0, stream>>>(
        h, hF, GAT_N * GAT_IN);
    transpose_W_kernel<<<dim3((GAT_IN * GAT_OUT + 255) / 256), dim3(256), 0, stream>>>(
        W, WT);
    zero_kernel<<<dim3((2 * GAT_H * GAT_N + 255) / 256), dim3(256), 0, stream>>>(
        siT, 2 * GAT_H * GAT_N);
    gemm1_kernel<<<dim3(128), dim3(256), 0, stream>>>(hF, WT, a, WhT, siT, sjT);
    gat_kernel<<<dim3(ntiles, S), dim3(32, 4), 0, stream>>>(
        adj, siT, sjT, WhT, out, Pacc, Pm, Pl, GAT_N / S, S);
    if (S > 1)
        combine_kernel<<<dim3(ntiles), dim3(64, 4), 0, stream>>>(
            Pacc, Pm, Pl, out, S);
}
